// Mamba2NoTriton_38422777430598
// MI455X (gfx1250) — matrix operand order repaired and checked by reading, not compiled, not run on hardware
//
#include <hip/hip_runtime.h>
#include <cstddef>
#include <cstdint>

// ---------------- model constants ----------------
#define DMODEL    2048
#define DSTATE    64
#define DCONV     4
#define HEADDIM   128
#define NHEADS    32
#define DINNER    4096
#define DINPROJ   8352           // 2*DINNER + 2*DSTATE + NHEADS
#define CONVDIM   4224           // DINNER + 2*DSTATE
#define CHUNKL    256
#define NCHUNK    8
#define RMS_EPS   1e-5f

typedef __attribute__((ext_vector_type(16))) __bf16 v16bf;
typedef __attribute__((ext_vector_type(8)))  float  v8f;

union FragU { v16bf v; unsigned u[8]; __bf16 h[16]; };

// Native converts: backend selects hardware bf16 cvt (packs to v_cvt_pk_*).
__device__ __forceinline__ __bf16 f2bf(float f) { return (__bf16)f; }
__device__ __forceinline__ float bf2f(__bf16 h) { return (float)h; }

__device__ __forceinline__ v8f wmma_bf16(v16bf a, v16bf b, v8f c) {
  return __builtin_amdgcn_wmma_f32_16x16x32_bf16(false, a, false, b, (short)0, c,
                                                 false, false);
}

// A fragment (16x32 MxK), layout: lane<16 -> K 0..15, lane>=16 -> K 16..31
__device__ __forceinline__ v16bf ldA(const __bf16* base, int strideh, int row,
                                     int kofs, int lane) {
  FragU f;
  const unsigned* p = (const unsigned*)base;
  int kb = kofs + ((lane & 16) ? 16 : 0);
  int o1 = (row * strideh + kb) >> 1;
#pragma unroll
  for (int j = 0; j < 8; ++j) f.u[j] = p[o1 + j];
  return f.v;
}
// B fragment (32x16 KxN), contiguous-k layout (same as ldA): lane<16 -> K 0..15, lane>=16 -> K 16..31
__device__ __forceinline__ v16bf ldB(const __bf16* base, int strideh, int col,
                                     int kofs, int lane) {
  FragU f;
  const unsigned* p = (const unsigned*)base;
  int o = (col * strideh + kofs + ((lane & 16) ? 16 : 0)) >> 1;
#pragma unroll
  for (int j = 0; j < 8; ++j) f.u[j] = p[o + j];
  return f.v;
}
// B fragment sourced from f32 LDS (converted on the fly)
__device__ __forceinline__ v16bf ldB_f32(const float* base, int stridef, int col,
                                         int kofs, int lane) {
  FragU f;
  const float* p = base + col * stridef + kofs + ((lane & 16) ? 16 : 0);
#pragma unroll
  for (int j = 0; j < 16; ++j) f.h[j] = f2bf(p[j]);
  return f.v;
}

__device__ __forceinline__ void load16(const float* g, float t[16]) {
  float4 r0 = ((const float4*)g)[0];
  float4 r1 = ((const float4*)g)[1];
  float4 r2 = ((const float4*)g)[2];
  float4 r3 = ((const float4*)g)[3];
  t[0] = r0.x;  t[1] = r0.y;  t[2] = r0.z;  t[3] = r0.w;
  t[4] = r1.x;  t[5] = r1.y;  t[6] = r1.z;  t[7] = r1.w;
  t[8] = r2.x;  t[9] = r2.y;  t[10] = r2.z; t[11] = r2.w;
  t[12] = r3.x; t[13] = r3.y; t[14] = r3.z; t[15] = r3.w;
}
__device__ __forceinline__ void stage16(__bf16* d, const float t[16]) {
#pragma unroll
  for (int j = 0; j < 16; ++j) d[j] = f2bf(t[j]);
}

// =====================================================================
// Kernel 1/6: C[M,N] = A[M,K] * B[N,K]^T   (fp32 in, bf16 WMMA, f32 out)
// 256 thr = 8 waves; tile 128x128x32; wave tile 32x64.
// Ping-pong double-buffered LDS, one barrier per K-step, next-tile
// register preload overlapping the WMMAs.
// =====================================================================
__global__ __launch_bounds__(256) void gemm_bf16_xt_kernel(
    const float* __restrict__ A, const float* __restrict__ B,
    float* __restrict__ C, int M, int N, int K) {
  __shared__ __bf16 As[2][128 * 40];
  __shared__ __bf16 Bs[2][128 * 40];
  const int tid = threadIdx.x;
  const int lane = tid & 31, wave = tid >> 5;
  const int wm = wave >> 1, wn = wave & 1;           // 4x2 wave grid
  const int m0 = blockIdx.y * 128, n0 = blockIdx.x * 128;
  const int lrow = tid >> 1, lcol = (tid & 1) * 16;  // staging map

  const float* aptr = A + (size_t)(m0 + lrow) * K + lcol;
  const bool bvalid = (n0 + lrow) < N;   // N tail guard (N=8352 % 128 != 0)
  const float* bptr = B + (size_t)(bvalid ? (n0 + lrow) : 0) * K + lcol;

  float ta[16], tb[16];
  // prologue: stage tile 0
  load16(aptr, ta);
  if (bvalid) load16(bptr, tb);
  else {
#pragma unroll
    for (int j = 0; j < 16; ++j) tb[j] = 0.f;
  }
  stage16(&As[0][lrow * 40 + lcol], ta);
  stage16(&Bs[0][lrow * 40 + lcol], tb);
  __syncthreads();

  v8f acc[2][4] = {};
  int p = 0;
  for (int k0 = 0; k0 < K; k0 += 32) {
    const bool has_next = (k0 + 32 < K);
    if (has_next) {  // issue next-tile global loads before compute
      if (k0 + 64 < K) __builtin_prefetch(aptr + k0 + 64, 0, 0);
      load16(aptr + k0 + 32, ta);
      if (bvalid) load16(bptr + k0 + 32, tb);
    }
    v16bf af[2], bfg[4];
#pragma unroll
    for (int mf = 0; mf < 2; ++mf)
      af[mf] = ldA(As[p], 40, wm * 32 + mf * 16 + (lane & 15), 0, lane);
#pragma unroll
    for (int nf = 0; nf < 4; ++nf)
      bfg[nf] = ldB(Bs[p], 40, wn * 64 + nf * 16 + (lane & 15), 0, lane);
#pragma unroll
    for (int mf = 0; mf < 2; ++mf)
#pragma unroll
      for (int nf = 0; nf < 4; ++nf)
        acc[mf][nf] = wmma_bf16(af[mf], bfg[nf], acc[mf][nf]);
    if (has_next) {
      stage16(&As[p ^ 1][lrow * 40 + lcol], ta);
      stage16(&Bs[p ^ 1][lrow * 40 + lcol], tb);
    }
    __syncthreads();
    p ^= 1;
  }
  // epilogue: D-frag layout lane<16 -> M=r, lane>=16 -> M=r+8 ; N = lane&15
#pragma unroll
  for (int mf = 0; mf < 2; ++mf)
#pragma unroll
    for (int nf = 0; nf < 4; ++nf)
#pragma unroll
      for (int r = 0; r < 8; ++r) {
        int row = m0 + wm * 32 + mf * 16 + ((lane & 16) ? r + 8 : r);
        int col = n0 + wn * 64 + nf * 16 + (lane & 15);
        if (col < N) C[(size_t)row * N + col] = acc[mf][nf][r];
      }
}

// =====================================================================
// Kernel 2/6: depthwise causal conv(4) + bias + SiLU over xBC slice
// =====================================================================
__global__ __launch_bounds__(256) void conv_silu_kernel(
    const float* __restrict__ zx, const float* __restrict__ wconv,
    const float* __restrict__ bconv, float* __restrict__ xbc, int Bsz, int L) {
  size_t idx = (size_t)blockIdx.x * 256 + threadIdx.x;
  size_t total = (size_t)Bsz * L * CONVDIM;
  if (idx >= total) return;
  int ch = (int)(idx % CONVDIM);
  size_t pos = idx / CONVDIM;
  int l = (int)(pos % L);
  int b = (int)(pos / L);
  float acc = bconv[ch];
#pragma unroll
  for (int t = 0; t < DCONV; ++t) {
    int ls = l - (DCONV - 1) + t;
    if (ls >= 0)
      acc += wconv[ch * DCONV + t] *
             zx[((size_t)b * L + ls) * DINPROJ + DINNER + ch];
  }
  acc = acc / (1.f + __expf(-acc));  // SiLU
  xbc[((size_t)b * L + l) * CONVDIM + ch] = acc;
}

// =====================================================================
// Kernel 3/6: dt = softplus(dt_raw + dt_bias)
// =====================================================================
__global__ __launch_bounds__(256) void dt_softplus_kernel(
    const float* __restrict__ zx, const float* __restrict__ dt_bias,
    float* __restrict__ dtsp, int Bsz, int L) {
  size_t idx = (size_t)blockIdx.x * 256 + threadIdx.x;
  size_t total = (size_t)Bsz * L * NHEADS;
  if (idx >= total) return;
  int h = (int)(idx % NHEADS);
  size_t tok = idx / NHEADS;
  float x = zx[tok * DINPROJ + DINNER + CONVDIM + h] + dt_bias[h];
  dtsp[idx] = (x > 20.f) ? x : log1pf(__expf(x));
}

// =====================================================================
// Kernel 4/6: SSD chunked scan. One workgroup per (batch, head); 8 waves.
// Per chunk: S=C·B^T (masked by exp(segsum)), Y_off=C·state·exp(acum),
// Y_diag=S'·X, chunk-state GEMM, state recurrence.
// =====================================================================
#define SSD_CS_OFF   0
#define SSD_BS_OFF   (32768)
#define SSD_BT_OFF   (65536)
#define SSD_XT_OFF   (98304)
#define SSD_SB_OFF   (163840)
#define SSD_ST_OFF   (198656)            // 163840 + 34816
#define SSD_ACUM_OFF (231424)
#define SSD_DECB_OFF (232448)
#define SSD_DTS_OFF  (233472)
#define SSD_SMEM_BYTES (234496)

__global__ __launch_bounds__(256) void ssd_chunk_kernel(
    const float* __restrict__ xbc, const float* __restrict__ dtsp,
    const float* __restrict__ A_log, float* __restrict__ Yb, int Bsz, int L) {
  extern __shared__ char smem[];
  __bf16* Cs   = (__bf16*)(smem + SSD_CS_OFF);   // [256][64]  C chunk
  __bf16* Bs   = (__bf16*)(smem + SSD_BS_OFF);   // [256][64]  B chunk
  __bf16* Bt   = (__bf16*)(smem + SSD_BT_OFF);   // [64][256]  B^T
  __bf16* Xt   = (__bf16*)(smem + SSD_XT_OFF);   // [128][256] (dt*x)^T
  __bf16* Sbuf = (__bf16*)(smem + SSD_SB_OFF);   // [64][272]  masked S block
  float*  St   = (float*)(smem + SSD_ST_OFF);    // [128][64]  running state^T
  float*  acum = (float*)(smem + SSD_ACUM_OFF);  // [256] cumsum(A*dt)
  float*  decB = (float*)(smem + SSD_DECB_OFF);  // [256] exp(alast-acum)
  float*  dts  = (float*)(smem + SSD_DTS_OFF);   // [256] dt

  const int bh = blockIdx.x;
  const int b = bh >> 5, h = bh & 31;
  const int tid = threadIdx.x;
  const int lane = tid & 31, wave = tid >> 5;
  const int wm4 = wave >> 1, wn2 = wave & 1;
  const float Ah = -__expf(A_log[h]);

  for (int i = tid; i < HEADDIM * DSTATE; i += 256) St[i] = 0.f;

  for (int c = 0; c < NCHUNK; ++c) {
    const int l0 = c * CHUNKL;
    {  // seed dt and A*dt
      float dtv = dtsp[((size_t)(b * L + l0 + tid)) * NHEADS + h];
      dts[tid] = dtv;
      acum[tid] = Ah * dtv;
    }
    __syncthreads();
    // inclusive Hillis-Steele scan of acum
    float v = acum[tid];
#pragma unroll
    for (int off = 1; off < 256; off <<= 1) {
      float add = (tid >= off) ? acum[tid - off] : 0.f;
      __syncthreads();
      v += add;
      acum[tid] = v;
      __syncthreads();
    }
    const float alast = acum[CHUNKL - 1];
    decB[tid] = __expf(alast - acum[tid]);
    // stage B and C chunks (NGROUPS==1: shared across heads)
    for (int i = tid; i < CHUNKL * DSTATE; i += 256) {
      int l = i >> 6, n = i & 63;
      size_t base = ((size_t)(b * L + l0 + l)) * CONVDIM + DINNER;
      __bf16 bb = f2bf(xbc[base + n]);
      Bs[l * 64 + n] = bb;
      Bt[n * 256 + l] = bb;
      Cs[l * 64 + n] = f2bf(xbc[base + DSTATE + n]);
    }
    // stage X (dt-scaled), transposed
    for (int i = tid; i < CHUNKL * HEADDIM; i += 256) {
      int l = i >> 7, p = i & 127;
      float xv = xbc[((size_t)(b * L + l0 + l)) * CONVDIM + h * HEADDIM + p];
      Xt[p * 256 + l] = f2bf(xv * dts[l]);
    }
    __syncthreads();

    // ---- process 4 row blocks of 64 ----
    for (int rb = 0; rb < 4; ++rb) {
      // (1) S block = C_rb (64x64) @ B^T (64x256); wave: 1x8 frags
      v8f s[8] = {};
#pragma unroll
      for (int ks = 0; ks < DSTATE; ks += 32) {
        v16bf a = ldA(Cs, 64, rb * 64 + wm4 * 16 + (lane & 15), ks, lane);
#pragma unroll
        for (int nf = 0; nf < 8; ++nf) {
          v16bf bb = ldB(Bs, 64, wn2 * 128 + nf * 16 + (lane & 15), ks, lane);
          s[nf] = wmma_bf16(a, bb, s[nf]);
        }
      }
      // mask by causal decay exp(acum[i]-acum[j]) and stage as bf16 A operand
#pragma unroll
      for (int r = 0; r < 8; ++r) {
        int ir = wm4 * 16 + ((lane & 16) ? r + 8 : r);
        int i = rb * 64 + ir;
        float ai = acum[i];
#pragma unroll
        for (int nf = 0; nf < 8; ++nf) {
          int j = wn2 * 128 + nf * 16 + (lane & 15);
          float val = (j <= i) ? s[nf][r] * __expf(ai - acum[j]) : 0.f;
          Sbuf[ir * 272 + j] = f2bf(val);
        }
      }
      __syncthreads();

      // (2) Y_rb = exp(acum)*(C_rb @ state_in) + S' @ X ; wave: 1x4 frags
      v8f y[4] = {};
#pragma unroll
      for (int ks = 0; ks < DSTATE; ks += 32) {
        v16bf a = ldA(Cs, 64, rb * 64 + wm4 * 16 + (lane & 15), ks, lane);
#pragma unroll
        for (int nf = 0; nf < 4; ++nf) {
          v16bf bb = ldB_f32(St, 64, wn2 * 64 + nf * 16 + (lane & 15), ks, lane);
          y[nf] = wmma_bf16(a, bb, y[nf]);
        }
      }
#pragma unroll
      for (int r = 0; r < 8; ++r) {
        int i = rb * 64 + wm4 * 16 + ((lane & 16) ? r + 8 : r);
        float e = __expf(acum[i]);
#pragma unroll
        for (int nf = 0; nf < 4; ++nf) y[nf][r] *= e;
      }
#pragma unroll
      for (int ks = 0; ks < CHUNKL; ks += 32) {
        v16bf a = ldA(Sbuf, 272, wm4 * 16 + (lane & 15), ks, lane);
#pragma unroll
        for (int nf = 0; nf < 4; ++nf) {
          v16bf bb = ldB(Xt, 256, wn2 * 64 + nf * 16 + (lane & 15), ks, lane);
          y[nf] = wmma_bf16(a, bb, y[nf]);
        }
      }
#pragma unroll
      for (int nf = 0; nf < 4; ++nf)
#pragma unroll
        for (int r = 0; r < 8; ++r) {
          int i = rb * 64 + wm4 * 16 + ((lane & 16) ? r + 8 : r);
          int p = wn2 * 64 + nf * 16 + (lane & 15);
          Yb[((size_t)(b * L + l0 + i) * NHEADS + h) * HEADDIM + p] = y[nf][r];
        }
      __syncthreads();
    }

    // (3) chunk states: cs[p][n] = sum_l (X[l][p]*decB[l]) * B[l][n]
    v8f cs[4] = {};
#pragma unroll
    for (int ks = 0; ks < CHUNKL; ks += 32) {
      FragU fa;
      int m = wave * 16 + (lane & 15);
      int kb = ((lane & 16) ? 16 : 0) + ks;
#pragma unroll
      for (int jj = 0; jj < 8; ++jj) {
        int k1 = kb + jj, k2 = kb + 8 + jj;
        fa.h[jj]     = f2bf(bf2f(Xt[m * 256 + k1]) * decB[k1]);
        fa.h[8 + jj] = f2bf(bf2f(Xt[m * 256 + k2]) * decB[k2]);
      }
#pragma unroll
      for (int nf = 0; nf < 4; ++nf) {
        v16bf bb = ldB(Bt, 256, nf * 16 + (lane & 15), ks, lane);
        cs[nf] = wmma_bf16(fa.v, bb, cs[nf]);
      }
    }
    __syncthreads();
    // state recurrence: St = exp(alast)*St + cs (each (p,n) owned by one lane)
    const float cdec = __expf(alast);
#pragma unroll
    for (int nf = 0; nf < 4; ++nf)
#pragma unroll
      for (int r = 0; r < 8; ++r) {
        int p = wave * 16 + ((lane & 16) ? r + 8 : r);
        int n = nf * 16 + (lane & 15);
        St[p * 64 + n] = cdec * St[p * 64 + n] + cs[nf][r];
      }
    __syncthreads();
  }
}

// =====================================================================
// Kernel 5/6: y = Y + x*D[h]; g = y*silu(z); RMSNorm(g)*norm_w  (per token)
// =====================================================================
__global__ __launch_bounds__(256) void gate_norm_kernel(
    const float* __restrict__ zx, const float* __restrict__ xbc,
    const float* __restrict__ Yb, const float* __restrict__ Dp,
    const float* __restrict__ normw, float* __restrict__ G) {
  const size_t tok = blockIdx.x;
  __shared__ float red[8];
  const float* z = zx + tok * DINPROJ;
  const float* xr = xbc + tok * CONVDIM;
  const float* yr = Yb + tok * DINNER;
  const int tid = threadIdx.x;
  float gv[16];
  float ss = 0.f;
#pragma unroll
  for (int j = 0; j < 16; ++j) {
    int i = tid + j * 256;
    float yv = yr[i] + xr[i] * Dp[i >> 7];
    float zv = z[i];
    float g = yv * (zv / (1.f + __expf(-zv)));
    gv[j] = g;
    ss += g * g;
  }
#pragma unroll
  for (int off = 16; off > 0; off >>= 1) ss += __shfl_down(ss, off);
  if ((tid & 31) == 0) red[tid >> 5] = ss;
  __syncthreads();
  float tot = 0.f;
#pragma unroll
  for (int w = 0; w < 8; ++w) tot += red[w];
  const float rs = rsqrtf(tot * (1.f / DINNER) + RMS_EPS);
#pragma unroll
  for (int j = 0; j < 16; ++j) {
    int i = tid + j * 256;
    G[tok * DINNER + i] = gv[j] * rs * normw[i];
  }
}

// =====================================================================
// Launch
// =====================================================================
extern "C" void kernel_launch(void* const* d_in, const int* in_sizes, int n_in,
                              void* d_out, int out_size, void* d_ws,
                              size_t ws_size, hipStream_t stream) {
  (void)in_sizes; (void)n_in; (void)out_size; (void)ws_size;
  const float* u       = (const float*)d_in[0];
  const float* W_in    = (const float*)d_in[1];
  const float* W_conv  = (const float*)d_in[2];
  const float* b_conv  = (const float*)d_in[3];
  const float* dt_bias = (const float*)d_in[4];
  const float* A_log   = (const float*)d_in[5];
  const float* Dp      = (const float*)d_in[6];
  const float* norm_w  = (const float*)d_in[7];
  const float* W_out   = (const float*)d_in[8];
  float* out = (float*)d_out;

  const int Bsz = 2, L = 2048;
  const int M = Bsz * L;  // 4096 tokens

  float* zx   = (float*)d_ws;                        // M x 8352
  float* xbc  = zx   + (size_t)M * DINPROJ;          // M x 4224
  float* dtsp = xbc  + (size_t)M * CONVDIM;          // M x 32
  float* Yb   = dtsp + (size_t)M * NHEADS;           // M x 4096
  float* G    = Yb   + (size_t)M * DINNER;           // M x 4096

  dim3 blk(256);

  // 1) in-projection: zx = u @ W_in^T
  dim3 g1((DINPROJ + 127) / 128, M / 128);
  gemm_bf16_xt_kernel<<<g1, blk, 0, stream>>>(u, W_in, zx, M, DINPROJ, DMODEL);

  // 2) depthwise conv + SiLU
  size_t convTot = (size_t)M * CONVDIM;
  conv_silu_kernel<<<(unsigned)((convTot + 255) / 256), blk, 0, stream>>>(
      zx, W_conv, b_conv, xbc, Bsz, L);

  // 3) dt softplus
  size_t dtTot = (size_t)M * NHEADS;
  dt_softplus_kernel<<<(unsigned)((dtTot + 255) / 256), blk, 0, stream>>>(
      zx, dt_bias, dtsp, Bsz, L);

  // 4) SSD chunked scan (64 WGs: one per batch*head)
  ssd_chunk_kernel<<<Bsz * NHEADS, blk, SSD_SMEM_BYTES, stream>>>(
      xbc, dtsp, A_log, Yb, Bsz, L);

  // 5) gate + RMSNorm
  gate_norm_kernel<<<M, blk, 0, stream>>>(zx, xbc, Yb, Dp, norm_w, G);

  // 6) out-projection: out = G @ W_out^T
  dim3 g2(DMODEL / 128, M / 128);
  gemm_bf16_xt_kernel<<<g2, blk, 0, stream>>>(G, W_out, out, M, DMODEL, DINNER);
}
